// CrossAttentionInjection_66073776881771
// MI455X (gfx1250) — matrix operand order repaired and checked by reading, not compiled, not run on hardware
//
#include <hip/hip_runtime.h>
#include <hip/hip_bf16.h>

// ---------------------------------------------------------------------------
// CrossAttentionInjection for MI455X (gfx1250, wave32, WMMA bf16)
// B=4, T=2048 (8 frames x 256 patches), Q_DIM=1024, KV_DIM=768,
// HEADS=16, DIM_HEAD=64, INNER=1024
// ---------------------------------------------------------------------------

typedef __bf16 v16bf __attribute__((ext_vector_type(16)));
typedef float  v8f   __attribute__((ext_vector_type(8)));

struct alignas(16) B128 { unsigned int u[4]; };

union FragB {
  v16bf v;
  B128  b[2];
  unsigned short s[16];
};

__device__ __forceinline__ unsigned short f2bf(float f) {
  unsigned int u = __float_as_uint(f);
  u += 0x7FFFu + ((u >> 16) & 1u);   // round-to-nearest-even
  return (unsigned short)(u >> 16);
}

// ---------------------------------------------------------------------------
// elementwise f32 -> bf16 cast (weights, context)
// ---------------------------------------------------------------------------
__global__ __launch_bounds__(256) void cast_bf16_kernel(
    const float* __restrict__ in, unsigned short* __restrict__ out, long long n) {
  long long i = (long long)blockIdx.x * blockDim.x + threadIdx.x;
  long long stride = (long long)gridDim.x * blockDim.x;
  for (; i < n; i += stride) out[i] = f2bf(in[i]);
}

// ---------------------------------------------------------------------------
// LayerNorm over 1024 cols, one block per row, output bf16
// ---------------------------------------------------------------------------
__global__ __launch_bounds__(256) void ln_kernel(
    const float* __restrict__ x, const float* __restrict__ g,
    const float* __restrict__ bta, unsigned short* __restrict__ xn) {
  __shared__ float red[256];
  const int row = blockIdx.x, t = threadIdx.x;
  const float* xr = x + (long long)row * 1024;
  float v[4];
#pragma unroll
  for (int i = 0; i < 4; i++) v[i] = xr[t + i * 256];
  float s = v[0] + v[1] + v[2] + v[3];
  red[t] = s; __syncthreads();
  for (int o = 128; o > 0; o >>= 1) { if (t < o) red[t] += red[t + o]; __syncthreads(); }
  float mean = red[0] * (1.0f / 1024.0f);
  __syncthreads();
  float s2 = 0.f;
#pragma unroll
  for (int i = 0; i < 4; i++) { float d = v[i] - mean; s2 += d * d; }
  red[t] = s2; __syncthreads();
  for (int o = 128; o > 0; o >>= 1) { if (t < o) red[t] += red[t + o]; __syncthreads(); }
  float var = red[0] * (1.0f / 1024.0f);
  float rs = rsqrtf(var + 1e-5f);
  unsigned short* xo = xn + (long long)row * 1024;
#pragma unroll
  for (int i = 0; i < 4; i++) {
    int c = t + i * 256;
    xo[c] = f2bf((v[i] - mean) * rs * g[c] + bta[c]);
  }
}

// ---------------------------------------------------------------------------
// Tiled bf16 GEMM: C[M,N] = scale * (A[M,K] @ B[K,N]) (+ bias[N])
// Block tile 128x128, BK=32, 256 threads = 8 waves, each wave 32x64 (2x4 WMMA)
// ---------------------------------------------------------------------------
template <bool OUT_BF16, bool HAS_BIAS>
__global__ __launch_bounds__(256) void gemm_bf16_kernel(
    const unsigned short* __restrict__ A, const unsigned short* __restrict__ Bm,
    float* __restrict__ Cf, unsigned short* __restrict__ Cbf,
    const float* __restrict__ bias, int M, int N, int K, float scale) {
  __shared__ unsigned short As[128 * 40];   // [row][k], stride 40 (16B-aligned rows)
  __shared__ unsigned short BsT[128 * 40];  // [n][k] transposed, stride 40
  const int t = threadIdx.x;
  const int lane = t & 31, wid = t >> 5;
  const int wm = wid & 3, wn = wid >> 2;
  const int m0 = blockIdx.y * 128, n0 = blockIdx.x * 128;

  v8f acc[2][4];
#pragma unroll
  for (int i = 0; i < 2; i++)
#pragma unroll
    for (int j = 0; j < 4; j++)
#pragma unroll
      for (int r = 0; r < 8; r++) acc[i][j][r] = 0.f;

  const int lrow = t >> 1, lkp = (t & 1) * 16;  // A-tile loader: 16 elems/thread
  const int brow = t >> 3, bnp = (t & 7) * 16;  // B-tile loader: 16 elems/thread

  for (int kb = 0; kb < K; kb += 32) {
    // stage A tile (row-major)
    const unsigned short* ag = A + (long long)(m0 + lrow) * K + kb + lkp;
    B128 a0 = *(const B128*)ag;
    B128 a1 = *(const B128*)(ag + 8);
    *(B128*)&As[lrow * 40 + lkp]     = a0;
    *(B128*)&As[lrow * 40 + lkp + 8] = a1;
    // stage B tile transposed -> contiguous-K fragment reads
    const unsigned short* bg = Bm + (long long)(kb + brow) * N + n0 + bnp;
    union { B128 b[2]; unsigned short s[16]; } tb;
    tb.b[0] = *(const B128*)bg;
    tb.b[1] = *(const B128*)(bg + 8);
#pragma unroll
    for (int i = 0; i < 16; i++) BsT[(bnp + i) * 40 + brow] = tb.s[i];

    if (kb + 32 < K) {
      __builtin_prefetch(A + (long long)(m0 + lrow) * K + kb + 32 + lkp, 0, 1);
      __builtin_prefetch(Bm + (long long)(kb + 32 + brow) * N + n0 + bnp, 0, 1);
    }
    __syncthreads();

    // A fragments: lane = M row (lane&15), K half selected by lane>>4
    FragB af[2];
#pragma unroll
    for (int mt = 0; mt < 2; mt++) {
      int r  = wm * 32 + mt * 16 + (lane & 15);
      int kc = (lane >> 4) * 8;
      af[mt].b[0] = *(const B128*)&As[r * 40 + kc];
      af[mt].b[1] = *(const B128*)&As[r * 40 + kc + 16];
    }
#pragma unroll
    for (int nt = 0; nt < 4; nt++) {
      // B fragment: lane = N col (lane&15), K half selected by lane>>4
      FragB bf;
      int nl = wn * 64 + nt * 16 + (lane & 15);
      int k0 = (lane >> 4) * 8;
      bf.b[0] = *(const B128*)&BsT[nl * 40 + k0];
      bf.b[1] = *(const B128*)&BsT[nl * 40 + k0 + 16];
#pragma unroll
      for (int mt = 0; mt < 2; mt++)
        acc[mt][nt] = __builtin_amdgcn_wmma_f32_16x16x32_bf16(
            false, af[mt].v, false, bf.v, (short)0, acc[mt][nt], false, false);
    }
    __syncthreads();
  }

  // epilogue: C/D layout: elem r of lane L -> M = r + (L>>4)*8, N = L&15
#pragma unroll
  for (int mt = 0; mt < 2; mt++)
#pragma unroll
    for (int nt = 0; nt < 4; nt++)
#pragma unroll
      for (int r = 0; r < 8; r++) {
        int row = m0 + wm * 32 + mt * 16 + (lane >> 4) * 8 + r;
        int col = n0 + wn * 64 + nt * 16 + (lane & 15);
        float v = acc[mt][nt][r] * scale;
        if (HAS_BIAS) v += bias[col];
        if (OUT_BF16) Cbf[(long long)row * N + col] = f2bf(v);
        else          Cf[(long long)row * N + col]  = v;
      }
}

// ---------------------------------------------------------------------------
// Attention: one block per (batch, head, frame). 512 threads = 16 waves.
// Each wave: 16x256 score strip (q pre-scaled by d^-0.5), exact softmax over
// the in-frame 256 cols (block-diagonal mask is structural), then P @ V.
// ALL accumulator-indexing loops are fully unrolled so sacc/oacc stay in
// VGPRs (partial unroll previously forced an alloca -> scratch spills).
// ---------------------------------------------------------------------------
__global__ __launch_bounds__(512) void attn_kernel(
    const unsigned short* __restrict__ qb,   // [8192,1024] bf16, pre-scaled
    const unsigned short* __restrict__ kvb,  // [8192,2048] bf16 (k | v)
    unsigned short* __restrict__ ob) {       // [8192,1024] bf16
  extern __shared__ unsigned short smem[];
  unsigned short* p  = smem;               // [256][264] probabilities (bf16)
  unsigned short* vT = smem + 256 * 264;   // [64][264]  V transposed (d-major)
  const int PSTR = 264;

  const int t = threadIdx.x, lane = t & 31, w = t >> 5;
  const int bid = blockIdx.x;
  const int b = bid >> 7;
  const int h = (bid >> 3) & 15;
  const int f = bid & 7;
  const long long rowbase = (long long)b * 2048 + f * 256;
  const int hoff = h * 64;
  const int vcol = 1024 + hoff;

  // stage V^T into LDS: vT[d][row]
#pragma unroll
  for (int it = 0; it < 4; ++it) {
    int task = t + it * 512;
    int row = task >> 3, dg = (task & 7) * 8;
    union { B128 b; unsigned short s[8]; } tv;
    tv.b = *(const B128*)&kvb[(rowbase + row) * 2048 + vcol + dg];
#pragma unroll
    for (int i = 0; i < 8; i++) vT[(dg + i) * PSTR + row] = tv.s[i];
  }
  __syncthreads();

  // ---- stage 1: S = q @ k^T for rows [w*16, w*16+16) ----
  FragB aq[2];
  {
    long long qrow = rowbase + w * 16 + (lane & 15);
#pragma unroll
    for (int kc = 0; kc < 2; kc++) {
      const unsigned short* src = qb + qrow * 1024 + hoff + kc * 32 + (lane >> 4) * 8;
      aq[kc].b[0] = *(const B128*)src;
      aq[kc].b[1] = *(const B128*)(src + 16);
    }
  }
  v8f sacc[16];
#pragma unroll
  for (int nt = 0; nt < 16; nt++)
#pragma unroll
    for (int r = 0; r < 8; r++) sacc[nt][r] = 0.f;
#pragma unroll
  for (int nt = 0; nt < 16; nt++) {
    long long krow = rowbase + nt * 16 + (lane & 15);
#pragma unroll
    for (int kc = 0; kc < 2; kc++) {
      FragB bk;  // B fragment of k^T: same K order as A
      const unsigned short* src = kvb + krow * 2048 + hoff + kc * 32 + (lane >> 4) * 8;
      bk.b[0] = *(const B128*)src;
      bk.b[1] = *(const B128*)(src + 16);
      sacc[nt] = __builtin_amdgcn_wmma_f32_16x16x32_bf16(
          false, aq[kc].v, false, bk.v, (short)0, sacc[nt], false, false);
    }
  }

  // ---- softmax: row r+(lane>>4)*8 lives in one 16-lane half ----
  float mx[8], sm[8];
#pragma unroll
  for (int r = 0; r < 8; r++) {
    float m = -3.0e38f;
#pragma unroll
    for (int nt = 0; nt < 16; nt++) m = fmaxf(m, sacc[nt][r]);
#pragma unroll
    for (int off = 8; off >= 1; off >>= 1) m = fmaxf(m, __shfl_xor(m, off, 16));
    mx[r] = m;
  }
#pragma unroll
  for (int r = 0; r < 8; r++) sm[r] = 0.f;
#pragma unroll
  for (int nt = 0; nt < 16; nt++)
#pragma unroll
    for (int r = 0; r < 8; r++) {
      float e = __expf(sacc[nt][r] - mx[r]);
      sacc[nt][r] = e;
      sm[r] += e;
    }
#pragma unroll
  for (int r = 0; r < 8; r++) {
    float s2 = sm[r];
#pragma unroll
    for (int off = 8; off >= 1; off >>= 1) s2 += __shfl_xor(s2, off, 16);
    sm[r] = 1.0f / s2;
  }

  // write P (bf16) to LDS in row-major form for A-fragment reloads
  const int rbase = w * 16 + (lane >> 4) * 8;
#pragma unroll
  for (int nt = 0; nt < 16; nt++)
#pragma unroll
    for (int r = 0; r < 8; r++)
      p[(rbase + r) * PSTR + nt * 16 + (lane & 15)] = f2bf(sacc[nt][r] * sm[r]);
  __syncthreads();

  // ---- stage 2: O = P @ V  (K = 256 in 8 chunks of 32) ----
  v8f oacc[4];
#pragma unroll
  for (int nt = 0; nt < 4; nt++)
#pragma unroll
    for (int r = 0; r < 8; r++) oacc[nt][r] = 0.f;
  const int prow = w * 16 + (lane & 15);
#pragma unroll
  for (int kc = 0; kc < 8; kc++) {
    FragB ap;
    const unsigned short* src = p + prow * PSTR + kc * 32 + (lane >> 4) * 8;
    ap.b[0] = *(const B128*)src;
    ap.b[1] = *(const B128*)(src + 16);
#pragma unroll
    for (int nt = 0; nt < 4; nt++) {
      FragB bv;  // from vT: same K order as A
      const unsigned short* vs = vT + (nt * 16 + (lane & 15)) * PSTR + kc * 32 + (lane >> 4) * 8;
      bv.b[0] = *(const B128*)vs;
      bv.b[1] = *(const B128*)(vs + 16);
      oacc[nt] = __builtin_amdgcn_wmma_f32_16x16x32_bf16(
          false, ap.v, false, bv.v, (short)0, oacc[nt], false, false);
    }
  }

  // write attention output (bf16) in [B*T, INNER] layout
#pragma unroll
  for (int nt = 0; nt < 4; nt++)
#pragma unroll
    for (int r = 0; r < 8; r++) {
      long long row = rowbase + rbase + r;
      ob[row * 1024 + hoff + nt * 16 + (lane & 15)] = f2bf(oacc[nt][r]);
    }
}

// ---------------------------------------------------------------------------
// launcher
// ---------------------------------------------------------------------------
extern "C" void kernel_launch(void* const* d_in, const int* in_sizes, int n_in,
                              void* d_out, int out_size, void* d_ws, size_t ws_size,
                              hipStream_t stream) {
  (void)in_sizes; (void)n_in; (void)out_size; (void)ws_size;
  const float* x   = (const float*)d_in[0];
  const float* ctx = (const float*)d_in[1];
  const float* g   = (const float*)d_in[2];
  const float* bta = (const float*)d_in[3];
  const float* Wq  = (const float*)d_in[4];
  const float* Wkv = (const float*)d_in[5];
  const float* Wo  = (const float*)d_in[6];
  const float* bo  = (const float*)d_in[7];
  // d_in[8] = mask: block-diagonal over frames — handled structurally, unused.
  float* out = (float*)d_out;

  size_t off = 0;
  auto carve = [&](size_t bytes) -> unsigned short* {
    unsigned short* pp = (unsigned short*)((char*)d_ws + off);
    off += (bytes + 255) & ~(size_t)255;
    return pp;
  };
  unsigned short* xn   = carve((size_t)8192 * 1024 * 2);  // LN(x) bf16
  unsigned short* ctxb = carve((size_t)8192 * 768 * 2);   // context bf16
  unsigned short* wqb  = carve((size_t)1024 * 1024 * 2);
  unsigned short* wkvb = carve((size_t)768 * 2048 * 2);
  unsigned short* wob  = carve((size_t)1024 * 1024 * 2);
  unsigned short* qb   = carve((size_t)8192 * 1024 * 2);  // q * d^-0.5
  unsigned short* kvb  = carve((size_t)8192 * 2048 * 2);  // [k | v]
  unsigned short* aob  = carve((size_t)8192 * 1024 * 2);  // attention out

  cast_bf16_kernel<<<4096, 256, 0, stream>>>(ctx, ctxb, (long long)8192 * 768);
  cast_bf16_kernel<<<2048, 256, 0, stream>>>(Wq,  wqb,  (long long)1024 * 1024);
  cast_bf16_kernel<<<2048, 256, 0, stream>>>(Wkv, wkvb, (long long)768 * 2048);
  cast_bf16_kernel<<<2048, 256, 0, stream>>>(Wo,  wob,  (long long)1024 * 1024);
  ln_kernel<<<8192, 256, 0, stream>>>(x, g, bta, xn);

  dim3 blk(256);
  // q = LN(x) @ Wq, scale = DIM_HEAD^-0.5 folded in
  gemm_bf16_kernel<true, false><<<dim3(1024 / 128, 8192 / 128), blk, 0, stream>>>(
      xn, wqb, nullptr, qb, nullptr, 8192, 1024, 1024, 0.125f);
  // kv = context @ Wkv
  gemm_bf16_kernel<true, false><<<dim3(2048 / 128, 8192 / 128), blk, 0, stream>>>(
      ctxb, wkvb, nullptr, kvb, nullptr, 8192, 2048, 768, 1.0f);

  // attention: 4 batches * 16 heads * 8 frames = 512 blocks
  size_t shmem = (size_t)(256 * 264 + 64 * 264) * 2;
  attn_kernel<<<512, 512, shmem, stream>>>(qb, kvb, aob);

  // out = attn_out @ Wo + bo  (f32)
  gemm_bf16_kernel<false, true><<<dim3(1024 / 128, 8192 / 128), blk, 0, stream>>>(
      aob, wob, out, nullptr, bo, 8192, 1024, 1024, 1.0f);
}
